// ElectronGNNLayer_62113817035112
// MI455X (gfx1250) — matrix operand order repaired and checked by reading, not compiled, not run on hardware
//
#include <hip/hip_runtime.h>
#include <hip/hip_bf16.h>
#include <cstddef>

// ---------------------------------------------------------------------------
// ElectronGNNLayer for MI455X (gfx1250), wave32 + WMMA bf16.
//
//   1. hx_kernel      : hx0/hx1 = silu(E @ hWee[t] + b), hxne = silu(Nuc @ hWne + b)
//   2. edge_kernel x3 : fused  silu(feat@uW+b) -> silu(.@wW+b) * hx[sender]
//                       scattered into a per-block FULL z accumulator in LDS
//                       (ds_add_f32), flushed per-block with plain b128 stores
//                       (zero global atomics).
//   3. reduce_kernel x3: z[t] = sum over block partials (b128 streaming)
//   4. final_kernel   : silu(E@gWres+b) + sum_t silu(z[t]@gWz[t]+b) + E
// ---------------------------------------------------------------------------

typedef __attribute__((ext_vector_type(16))) __bf16 v16bf;
typedef __attribute__((ext_vector_type(8)))  float  v8f;

// Fast silu: v_exp_f32 + v_rcp_f32 instead of the IEEE divide expansion.
// (bf16-quantized GEMM inputs make 1-ulp rcp error irrelevant.)
__device__ __forceinline__ float silu_f(float x) {
  return x * __builtin_amdgcn_rcpf(1.0f + __expf(-x));
}

__device__ __forceinline__ v8f wmma_bf16(v16bf a, v16bf b, v8f c) {
  // D = A(16x32 bf16) * B(32x16 bf16) + C(16x16 f32)
  return __builtin_amdgcn_wmma_f32_16x16x32_bf16(false, a, false, b,
                                                 (short)0, c, false, false);
}

// A fragment: 16x32 tile at (row0, kbase) of row-major f32 [rows x ld].
// Layout (ISA 16-bit A 16x32 table): lane = row (m = lane&15, h = lane>>4),
// elements 0..7  -> K = kbase + 8h + e ;  8..15 -> K = kbase + 16 + 8h + (e-8)
// Both 8-element halves are contiguous f32 -> two pairs of b128 loads.
__device__ __forceinline__ v16bf load_a_frag(const float* __restrict__ base,
                                             int row0, int ld, int kbase) {
  int lane = threadIdx.x & 31;
  int m = lane & 15, h = lane >> 4;
  const float* p = base + (size_t)(row0 + m) * ld + kbase + 8 * h;
  float4 q0 = *(const float4*)(p);
  float4 q1 = *(const float4*)(p + 4);
  float4 q2 = *(const float4*)(p + 16);
  float4 q3 = *(const float4*)(p + 20);
  v16bf a;
  a[0]  = (__bf16)q0.x; a[1]  = (__bf16)q0.y; a[2]  = (__bf16)q0.z; a[3]  = (__bf16)q0.w;
  a[4]  = (__bf16)q1.x; a[5]  = (__bf16)q1.y; a[6]  = (__bf16)q1.z; a[7]  = (__bf16)q1.w;
  a[8]  = (__bf16)q2.x; a[9]  = (__bf16)q2.y; a[10] = (__bf16)q2.z; a[11] = (__bf16)q2.w;
  a[12] = (__bf16)q3.x; a[13] = (__bf16)q3.y; a[14] = (__bf16)q3.z; a[15] = (__bf16)q3.w;
  return a;
}

// B fragment: 32x16 tile of row-major f32 weights [K x ld], cols col0..+15.
// Layout (ISA 16-bit table, matches A): lane = col (n = lane&15), e<8 -> K = kbase + 8h + e ; e>=8 -> K = kbase + 16 + 8h + (e-8).
// Strided along K -> inherently scalar loads; hoisted out of hot loops.
__device__ __forceinline__ v16bf load_b_frag(const float* __restrict__ W,
                                             int ld, int kbase, int col0) {
  int lane = threadIdx.x & 31;
  int n = lane & 15, h = lane >> 4;
  v16bf b;
#pragma unroll
  for (int e = 0; e < 16; ++e)
    b[e] = (__bf16)W[(size_t)(kbase + 8 * h + e + (e >= 8 ? 8 : 0)) * ld + col0 + n];
  return b;
}

// ---------------------------------------------------------------------------
// Kernel 1: node feature transforms (tiny: ~13 MFLOP)
// ---------------------------------------------------------------------------
__global__ void hx_kernel(const float* __restrict__ electrons,
                          const float* __restrict__ nuclei,
                          const float* __restrict__ hWee,   // [2][128][64]
                          const float* __restrict__ hbee,   // [2][64]
                          const float* __restrict__ hWne,   // [64][64]
                          const float* __restrict__ hbne,   // [64]
                          float* __restrict__ hx0,          // [768][64]
                          float* __restrict__ hx1,          // [768][64]
                          float* __restrict__ hxne)         // [32][64]
{
  int gw   = (int)((blockIdx.x * blockDim.x + threadIdx.x) >> 5);
  int lane = threadIdx.x & 31;
  int n = lane & 15, h = lane >> 4;
  if (gw >= 98) return;  // wave-uniform

  if (gw < 96) {
    int which = gw / 48;
    int rt    = gw % 48;
    const float* W = hWee + which * (128 * 64);
    float* dst = which ? hx1 : hx0;
    v16bf a[4];
#pragma unroll
    for (int kt = 0; kt < 4; ++kt)
      a[kt] = load_a_frag(electrons, rt * 16, 128, kt * 32);
#pragma unroll
    for (int nt = 0; nt < 4; ++nt) {
      v8f acc = {};
#pragma unroll
      for (int kt = 0; kt < 4; ++kt)
        acc = wmma_bf16(a[kt], load_b_frag(W, 64, kt * 32, nt * 16), acc);
      int col = nt * 16 + n;
      float bias = hbee[which * 64 + col];
#pragma unroll
      for (int r = 0; r < 8; ++r)
        dst[(size_t)(rt * 16 + r + 8 * h) * 64 + col] = silu_f(acc[r] + bias);
    }
  } else {
    int rt = gw - 96;  // 0..1
    v16bf a[2];
#pragma unroll
    for (int kt = 0; kt < 2; ++kt)
      a[kt] = load_a_frag(nuclei, rt * 16, 64, kt * 32);
#pragma unroll
    for (int nt = 0; nt < 4; ++nt) {
      v8f acc = {};
#pragma unroll
      for (int kt = 0; kt < 2; ++kt)
        acc = wmma_bf16(a[kt], load_b_frag(hWne, 64, kt * 32, nt * 16), acc);
      int col = nt * 16 + n;
      float bias = hbne[col];
#pragma unroll
      for (int r = 0; r < 8; ++r)
        hxne[(size_t)(rt * 16 + r + 8 * h) * 64 + col] = silu_f(acc[r] + bias);
    }
  }
}

// ---------------------------------------------------------------------------
// Kernel 2: fused edge MLP + LDS-resident scatter.
// Dynamic LDS: [0, 192KB)   full z accumulator (768 x 64 f32)
//              [192KB, +2KB/wave) bf16 staging tile for the C->A repack
// Block = 256 threads (8 waves), one block per WGP (213 KB of 320 KB LDS).
// ---------------------------------------------------------------------------
__global__ void edge_kernel(const float* __restrict__ feat,      // [n_tiles*16][32]
                            const int*   __restrict__ senders,
                            const int*   __restrict__ receivers,
                            const float* __restrict__ uW,        // [32][64]
                            const float* __restrict__ ub,        // [64]
                            const float* __restrict__ wW,        // [64][64]
                            const float* __restrict__ wb,        // [64]
                            const float* __restrict__ hx,        // [.][64]
                            float* __restrict__ partial,         // [gridDim][768*64]
                            int n_tiles)
{
  extern __shared__ float smem[];
  float* zacc = smem;  // 49152 f32
  const int tid  = threadIdx.x;
  const int wave = tid >> 5;
  const int lane = tid & 31;
  const int n = lane & 15, h = lane >> 4;
  __bf16* stage = (__bf16*)(smem + 49152) + wave * 1024;  // 16x64 bf16 per wave

  {  // vectorized zero of the z accumulator (ds_store_b128)
    float4 zero4 = {0.f, 0.f, 0.f, 0.f};
    float4* z4 = (float4*)zacc;
    for (int i = tid; i < 49152 / 4; i += blockDim.x) z4[i] = zero4;
  }
  __syncthreads();

  // Resident weight fragments (per wave): 4 + 8 fragments = 96 VGPRs.
  v16bf Bu[4], Bw[2][4];
#pragma unroll
  for (int nt = 0; nt < 4; ++nt) Bu[nt] = load_b_frag(uW, 64, 0, nt * 16);
#pragma unroll
  for (int kt = 0; kt < 2; ++kt)
#pragma unroll
    for (int nt = 0; nt < 4; ++nt)
      Bw[kt][nt] = load_b_frag(wW, 64, kt * 32, nt * 16);
  float ubias[4], wbias[4];
#pragma unroll
  for (int nt = 0; nt < 4; ++nt) {
    ubias[nt] = ub[nt * 16 + n];
    wbias[nt] = wb[nt * 16 + n];
  }

  const int gw = (int)blockIdx.x * (blockDim.x >> 5) + wave;
  const int nw = (int)gridDim.x * (blockDim.x >> 5);

  for (int tile = gw; tile < n_tiles; tile += nw) {  // wave-uniform loop
    const int base = tile * 16;

    // Prefetch the next tile's feature rows (2 KB) into cache.
    if (tile + nw < n_tiles)
      __builtin_prefetch(feat + (size_t)(tile + nw) * 512 + lane * 16, 0, 1);

    // ---- layer 1: e = silu(feat @ uW + ub), K = 32 in ONE wmma per col tile
    v16bf a = load_a_frag(feat, base, 32, 0);
#pragma unroll
    for (int nt = 0; nt < 4; ++nt) {
      v8f c = {};
      c = wmma_bf16(a, Bu[nt], c);
#pragma unroll
      for (int r = 0; r < 8; ++r) {
        int m = r + 8 * h;  // C layout: lane=col, vgpr r -> row r + 8h
        stage[m * 64 + nt * 16 + n] = (__bf16)silu_f(c[r] + ubias[nt]);
      }
    }

    // ---- repack C layout -> A layout via per-wave LDS tile (same-wave deps)
    v16bf a2[2];
#pragma unroll
    for (int kt = 0; kt < 2; ++kt) {
      const __bf16* sp = stage + (lane & 15) * 64 + kt * 32;
      v16bf t;
#pragma unroll
      for (int e = 0; e < 8; ++e) t[e] = sp[8 * h + e];
#pragma unroll
      for (int e = 0; e < 8; ++e) t[8 + e] = sp[16 + 8 * h + e];
      a2[kt] = t;
    }

    // ---- per-row graph indices
    int srow[8], rrow[8];
#pragma unroll
    for (int r = 0; r < 8; ++r) {
      int m = r + 8 * h;
      srow[r] = senders[base + m];
      rrow[r] = receivers[base + m];
    }

    // ---- layer 2 + gather-multiply + LDS scatter-add
#pragma unroll
    for (int nt = 0; nt < 4; ++nt) {
      v8f c = {};
      c = wmma_bf16(a2[0], Bw[0][nt], c);
      c = wmma_bf16(a2[1], Bw[1][nt], c);  // chained f32 accumulate, K=64
      int col = nt * 16 + n;
#pragma unroll
      for (int r = 0; r < 8; ++r) {
        float we  = silu_f(c[r] + wbias[nt]);
        float msg = we * hx[(size_t)srow[r] * 64 + col];
        atomicAdd(&zacc[rrow[r] * 64 + col], msg);  // ds_add_f32
      }
    }
  }

  __syncthreads();
  // Flush block-private z with plain b128 streaming stores (no global atomics).
  {
    const float4* z4 = (const float4*)zacc;
    float4* out4 = (float4*)(partial + (size_t)blockIdx.x * 49152);
    for (int i = tid; i < 49152 / 4; i += blockDim.x) out4[i] = z4[i];
  }
}

// ---------------------------------------------------------------------------
// Kernel 3: sum per-block partials into z[t] (b128 streaming, ~14 MB total)
// grid 48 x 256 threads covers 12288 float4 = 49152 floats exactly.
// ---------------------------------------------------------------------------
__global__ void reduce_kernel(const float* __restrict__ partial,
                              float* __restrict__ z, int nblocks) {
  int i = (int)(blockIdx.x * blockDim.x + threadIdx.x);  // float4 index
  const float4* p4 = (const float4*)partial;
  float4 s = {0.f, 0.f, 0.f, 0.f};
  for (int b = 0; b < nblocks; ++b) {
    float4 v = p4[(size_t)b * (49152 / 4) + i];
    s.x += v.x; s.y += v.y; s.z += v.z; s.w += v.w;
  }
  ((float4*)z)[i] = s;
}

// ---------------------------------------------------------------------------
// Kernel 4: featurewise update + residual. One 16x16 output tile per wave.
// ---------------------------------------------------------------------------
__global__ void final_kernel(const float* __restrict__ electrons,  // [768][128]
                             const float* __restrict__ z0,
                             const float* __restrict__ z1,
                             const float* __restrict__ z2,
                             const float* __restrict__ gWres,      // [128][128]
                             const float* __restrict__ gbres,      // [128]
                             const float* __restrict__ gWz,        // [3][64][128]
                             const float* __restrict__ gbz,        // [3][128]
                             float* __restrict__ out)              // [768][128]
{
  int gw   = (int)((blockIdx.x * blockDim.x + threadIdx.x) >> 5);
  int lane = threadIdx.x & 31;
  int n = lane & 15, h = lane >> 4;
  if (gw >= 384) return;  // wave-uniform
  int rt = gw >> 3, nt = gw & 7;
  int row0 = rt * 16, col = nt * 16 + n;

  v8f acc = {};
#pragma unroll
  for (int kt = 0; kt < 4; ++kt)
    acc = wmma_bf16(load_a_frag(electrons, row0, 128, kt * 32),
                    load_b_frag(gWres, 128, kt * 32, nt * 16), acc);
  float res[8];
  float b0 = gbres[col];
#pragma unroll
  for (int r = 0; r < 8; ++r) res[r] = silu_f(acc[r] + b0);

  const float* zs[3] = {z0, z1, z2};
#pragma unroll
  for (int t = 0; t < 3; ++t) {
    v8f az = {};
#pragma unroll
    for (int kt = 0; kt < 2; ++kt)
      az = wmma_bf16(load_a_frag(zs[t], row0, 64, kt * 32),
                     load_b_frag(gWz + (size_t)t * 64 * 128, 128, kt * 32, nt * 16),
                     az);
    float bz = gbz[t * 128 + col];
#pragma unroll
    for (int r = 0; r < 8; ++r) res[r] += silu_f(az[r] + bz);
  }

#pragma unroll
  for (int r = 0; r < 8; ++r) {
    int m = r + 8 * h;
    out[(size_t)(row0 + m) * 128 + col] =
        res[r] + electrons[(size_t)(row0 + m) * 128 + col];
  }
}

// ---------------------------------------------------------------------------
extern "C" void kernel_launch(void* const* d_in, const int* in_sizes, int n_in,
                              void* d_out, int out_size, void* d_ws, size_t ws_size,
                              hipStream_t stream) {
  (void)in_sizes; (void)n_in; (void)out_size; (void)ws_size;
  const float* electrons      = (const float*)d_in[0];
  const float* nuclei         = (const float*)d_in[1];
  const float* feat_same      = (const float*)d_in[2];
  const float* feat_anti      = (const float*)d_in[3];
  const float* feat_ne        = (const float*)d_in[4];
  const int*   senders_same   = (const int*)d_in[5];
  const int*   receivers_same = (const int*)d_in[6];
  const int*   senders_anti   = (const int*)d_in[7];
  const int*   receivers_anti = (const int*)d_in[8];
  const int*   senders_ne     = (const int*)d_in[9];
  const int*   receivers_ne   = (const int*)d_in[10];
  const float* u_W    = (const float*)d_in[11];  // [3][32][64]
  const float* u_b    = (const float*)d_in[12];  // [3][64]
  const float* w_W    = (const float*)d_in[13];  // [3][64][64]
  const float* w_b    = (const float*)d_in[14];  // [3][64]
  const float* h_W_ee = (const float*)d_in[15];  // [2][128][64]
  const float* h_b_ee = (const float*)d_in[16];  // [2][64]
  const float* h_W_ne = (const float*)d_in[17];  // [64][64]
  const float* h_b_ne = (const float*)d_in[18];  // [64]
  const float* gWres  = (const float*)d_in[19];  // [128][128]
  const float* gbres  = (const float*)d_in[20];  // [128]
  const float* gWz    = (const float*)d_in[21];  // [3][64][128]
  const float* gbz    = (const float*)d_in[22];  // [3][128]

  // Workspace layout (floats): ~14.4 MB total
  float* ws   = (float*)d_ws;
  float* hx0  = ws;                    // 49152
  float* hx1  = ws + 49152;            // 49152
  float* hxne = ws + 98304;            // 2048
  float* z0   = ws + 100352;           // 49152
  float* z1   = ws + 149504;           // 49152
  float* z2   = ws + 198656;           // 49152
  float* p0   = ws + 247808;           // 32 * 49152
  float* p1   = p0 + (size_t)32 * 49152;
  float* p2   = p1 + (size_t)32 * 49152;  // 4 * 49152

  // 1) node transforms
  hx_kernel<<<13, 256, 0, stream>>>(electrons, nuclei, h_W_ee, h_b_ee,
                                    h_W_ne, h_b_ne, hx0, hx1, hxne);

  // 2) fused edge pipelines: 213 KB dynamic LDS (192 KB z + 8 x 2 KB staging)
  const size_t shmem = 49152 * sizeof(float) + 8 * 1024 * sizeof(__bf16);
  (void)hipFuncSetAttribute((const void*)edge_kernel,
                            hipFuncAttributeMaxDynamicSharedMemorySize,
                            (int)shmem);
  edge_kernel<<<32, 256, shmem, stream>>>(
      feat_same, senders_same, receivers_same,
      u_W + 0 * 2048, u_b + 0 * 64, w_W + 0 * 4096, w_b + 0 * 64,
      hx0, p0, 294144 / 16);
  edge_kernel<<<32, 256, shmem, stream>>>(
      feat_anti, senders_anti, receivers_anti,
      u_W + 1 * 2048, u_b + 1 * 64, w_W + 1 * 4096, w_b + 1 * 64,
      hx1, p1, 294912 / 16);
  edge_kernel<<<4, 256, shmem, stream>>>(
      feat_ne, senders_ne, receivers_ne,
      u_W + 2 * 2048, u_b + 2 * 64, w_W + 2 * 4096, w_b + 2 * 64,
      hxne, p2, 24576 / 16);

  // 3) partial reductions (float4 per thread)
  reduce_kernel<<<48, 256, 0, stream>>>(p0, z0, 32);
  reduce_kernel<<<48, 256, 0, stream>>>(p1, z1, 32);
  reduce_kernel<<<48, 256, 0, stream>>>(p2, z2, 4);

  // 4) featurewise update + residual
  final_kernel<<<48, 256, 0, stream>>>(electrons, z0, z1, z2,
                                       gWres, gbres, gWz, gbz, (float*)d_out);
}
